// Attention_17995912970360
// MI455X (gfx1250) — matrix operand order repaired and checked by reading, not compiled, not run on hardware
//
#include <hip/hip_runtime.h>
#include <hip/hip_bf16.h>

typedef __attribute__((ext_vector_type(16))) __bf16 v16bf;
typedef __attribute__((ext_vector_type(8)))  __bf16 v8bf;
typedef __attribute__((ext_vector_type(8)))  float  v8f;

#define C_DIM 512
#define N_DIM 4096
#define BATCH 2

__device__ __forceinline__ __bf16 f2bf(float f) {
  union { float f; unsigned u; } v; v.f = f;
  unsigned r = v.u + 0x7FFFu + ((v.u >> 16) & 1u);  // round-to-nearest-even
  unsigned short h = (unsigned short)(r >> 16);
  return __builtin_bit_cast(__bf16, h);
}

// A-fragment (16x32, 16-bit): lane m=l%16 holds row m.
// K = half*16 + 0..15
__device__ __forceinline__ v16bf load_a32(const __bf16* row, int half) {
  v8bf lo = *(const v8bf*)(row + half * 16);
  v8bf hi = *(const v8bf*)(row + half * 16 + 8);
  v16bf a;
#pragma unroll
  for (int e = 0; e < 8; ++e) { a[e] = lo[e]; a[8 + e] = hi[e]; }
  return a;
}

// B-fragment (32x16, 16-bit): lane n=l%16 holds col n, K = half*16 + 0..15
__device__ __forceinline__ v16bf load_b32(const __bf16* col, int half) {
  return *(const v16bf*)(col + half * 16);
}

__global__ void convert_f32_bf16(const float* __restrict__ src,
                                 __bf16* __restrict__ dst, int n) {
  int i = blockIdx.x * blockDim.x + threadIdx.x;
  int stride = gridDim.x * blockDim.x;
  for (; i < n; i += stride) dst[i] = f2bf(src[i]);
}

// One block per (group, batch). Writes normalized activation transposed:
// hnT[b][n][c] in bf16 so downstream GEMM B-frags read contiguous K=c.
__global__ __launch_bounds__(256) void groupnorm_kernel(
    const float* __restrict__ x, const float* __restrict__ gamma,
    const float* __restrict__ beta, __bf16* __restrict__ hnT) {
  const int g = blockIdx.x;   // 0..31
  const int b = blockIdx.y;   // 0..1
  const float* base = x + ((size_t)b * C_DIM + g * 16) * N_DIM;
  const int total = 16 * N_DIM;  // 65536
  float s = 0.f, s2 = 0.f;
  for (int i = threadIdx.x; i < total; i += blockDim.x) {
    float v = base[i]; s += v; s2 += v * v;
  }
#pragma unroll
  for (int off = 16; off >= 1; off >>= 1) {
    s += __shfl_xor(s, off); s2 += __shfl_xor(s2, off);
  }
  __shared__ float red[16];
  __shared__ float stats[2];
  int w = threadIdx.x >> 5;
  if ((threadIdx.x & 31) == 0) { red[w] = s; red[8 + w] = s2; }
  __syncthreads();
  if (threadIdx.x == 0) {
    float ts = 0.f, ts2 = 0.f;
    for (int i = 0; i < 8; ++i) { ts += red[i]; ts2 += red[8 + i]; }
    float mean = ts / (float)total;
    float var = ts2 / (float)total - mean * mean;
    stats[0] = mean;
    stats[1] = rsqrtf(var + 1e-6f);
  }
  __syncthreads();
  float mean = stats[0], inv = stats[1];
  for (int i = threadIdx.x; i < total; i += blockDim.x) {
    int cl = i >> 12;           // i / 4096
    int n = i & (N_DIM - 1);
    int c = g * 16 + cl;
    float v = (base[i] - mean) * inv * gamma[c] + beta[c];
    hnT[((size_t)b * N_DIM + n) * C_DIM + c] = f2bf(v);
  }
}

// C[o,n] = sum_c W[o,c]*X[c,n] + bias[o].  W bf16 [512][512], XT bf16 [B][4096][512].
// mode 0: store bf16 transposed outb[b][n][o]   (for q, k)
// mode 1: store bf16 outb[b][o][n]              (for v)
// mode 2: store f32 outf[b][o][n] = val + resid (for proj + residual)
__global__ __launch_bounds__(256) void gemm_bf16(
    const __bf16* __restrict__ W, const __bf16* __restrict__ XT,
    const float* __restrict__ bias, const float* __restrict__ resid,
    __bf16* __restrict__ outb, float* __restrict__ outf, int mode) {
  const int lane = threadIdx.x & 31;
  const int wave = threadIdx.x >> 5;
  const int half = lane >> 4;
  const int l16 = lane & 15;
  const int wo = wave & 1, wn = wave >> 1;
  const int b = blockIdx.z;
  const int oBase = blockIdx.y * 64 + wo * 32;
  const int nBase = blockIdx.x * 128 + wn * 32;
  const __bf16* xb = XT + (size_t)b * N_DIM * C_DIM;

  v8f acc[2][2] = {};
  for (int kc = 0; kc < C_DIM; kc += 32) {
    v16bf a0 = load_a32(W + (size_t)(oBase + l16) * C_DIM + kc, half);
    v16bf a1 = load_a32(W + (size_t)(oBase + 16 + l16) * C_DIM + kc, half);
    v16bf b0 = load_b32(xb + (size_t)(nBase + l16) * C_DIM + kc, half);
    v16bf b1 = load_b32(xb + (size_t)(nBase + 16 + l16) * C_DIM + kc, half);
    acc[0][0] = __builtin_amdgcn_wmma_f32_16x16x32_bf16(false, a0, false, b0, (short)0, acc[0][0], false, false);
    acc[0][1] = __builtin_amdgcn_wmma_f32_16x16x32_bf16(false, a0, false, b1, (short)0, acc[0][1], false, false);
    acc[1][0] = __builtin_amdgcn_wmma_f32_16x16x32_bf16(false, a1, false, b0, (short)0, acc[1][0], false, false);
    acc[1][1] = __builtin_amdgcn_wmma_f32_16x16x32_bf16(false, a1, false, b1, (short)0, acc[1][1], false, false);
  }
#pragma unroll
  for (int ot = 0; ot < 2; ++ot) {
#pragma unroll
    for (int nt = 0; nt < 2; ++nt) {
#pragma unroll
      for (int r = 0; r < 8; ++r) {
        int o = oBase + ot * 16 + r + 8 * half;   // D row
        int n = nBase + nt * 16 + l16;            // D col
        float val = acc[ot][nt][r] + bias[o];
        if (mode == 0) {
          outb[((size_t)b * N_DIM + n) * C_DIM + o] = f2bf(val);
        } else if (mode == 1) {
          outb[((size_t)b * C_DIM + o) * N_DIM + n] = f2bf(val);
        } else {
          size_t idx = ((size_t)b * C_DIM + o) * N_DIM + n;
          outf[idx] = val + resid[idx];
        }
      }
    }
  }
}

// Flash attention. 1 wave per block; block = (i-tile of 16 rows, c-half of 256, batch).
// qT/kT: [B][4096][512] bf16 ; v: [B][512][4096] bf16 ; oT out: [B][4096][512] bf16.
__global__ __launch_bounds__(32) void attn_kernel(
    const __bf16* __restrict__ qT, const __bf16* __restrict__ kT,
    const __bf16* __restrict__ vv, __bf16* __restrict__ oT) {
  const int lane = threadIdx.x;
  const int half = lane >> 4;
  const int l16 = lane & 15;
  const int iBase = blockIdx.x * 16;
  const int cBase0 = blockIdx.y * 256;
  const int b = blockIdx.z;
  const float scale = 0.04419417382415922f;  // 512^-0.5
  const __bf16* qb = qT + (size_t)b * N_DIM * C_DIM;
  const __bf16* kb = kT + (size_t)b * N_DIM * C_DIM;
  const __bf16* vb = vv + (size_t)b * C_DIM * N_DIM;

  float mrow[8], lrow[8];
#pragma unroll
  for (int r = 0; r < 8; ++r) { mrow[r] = -3.0e38f; lrow[r] = 0.f; }

  // ---- Pass 1: streaming row max / sum-exp (S = Q^T K, no store) ----
  for (int jt = 0; jt < N_DIM; jt += 16) {
    v8f s = {};
#pragma unroll 4
    for (int kc = 0; kc < C_DIM; kc += 32) {
      v16bf a  = load_a32(qb + (size_t)(iBase + l16) * C_DIM + kc, half);
      v16bf bf = load_b32(kb + (size_t)(jt + l16) * C_DIM + kc, half);
      s = __builtin_amdgcn_wmma_f32_16x16x32_bf16(false, a, false, bf, (short)0, s, false, false);
    }
#pragma unroll
    for (int r = 0; r < 8; ++r) {
      float v = s[r] * scale;
      float rm = v;
#pragma unroll
      for (int off = 1; off < 16; off <<= 1) rm = fmaxf(rm, __shfl_xor(rm, off, 16));
      float mnew = fmaxf(mrow[r], rm);
      float e = __expf(v - mnew);
#pragma unroll
      for (int off = 1; off < 16; off <<= 1) e += __shfl_xor(e, off, 16);
      lrow[r] = lrow[r] * __expf(mrow[r] - mnew) + e;
      mrow[r] = mnew;
    }
  }

  // ---- Pass 2: recompute S per 32-wide j-tile, P = exp(S-m), O += P @ V^T ----
  __shared__ __bf16 Plds[16 * 32];
  v8f oacc[16] = {};
  for (int jt = 0; jt < N_DIM; jt += 32) {
    v8f s0 = {}, s1 = {};
#pragma unroll 4
    for (int kc = 0; kc < C_DIM; kc += 32) {
      v16bf a  = load_a32(qb + (size_t)(iBase + l16) * C_DIM + kc, half);
      v16bf b0 = load_b32(kb + (size_t)(jt + l16) * C_DIM + kc, half);
      v16bf b1 = load_b32(kb + (size_t)(jt + 16 + l16) * C_DIM + kc, half);
      s0 = __builtin_amdgcn_wmma_f32_16x16x32_bf16(false, a, false, b0, (short)0, s0, false, false);
      s1 = __builtin_amdgcn_wmma_f32_16x16x32_bf16(false, a, false, b1, (short)0, s1, false, false);
    }
    // D layout -> LDS (row r+8*half, col l16); same-wave DS ops are in-order.
#pragma unroll
    for (int r = 0; r < 8; ++r) {
      int row = r + 8 * half;
      Plds[row * 32 + l16]      = f2bf(__expf(s0[r] * scale - mrow[r]));
      Plds[row * 32 + 16 + l16] = f2bf(__expf(s1[r] * scale - mrow[r]));
    }
    // Read P back as a 16x32 A-fragment (M=i rows, K=j)
    v16bf pA;
    {
      v8bf lo = *(const v8bf*)(&Plds[l16 * 32 + half * 16]);
      v8bf hi = *(const v8bf*)(&Plds[l16 * 32 + half * 16 + 8]);
#pragma unroll
      for (int e = 0; e < 8; ++e) { pA[e] = lo[e]; pA[8 + e] = hi[e]; }
    }
#pragma unroll
    for (int ct = 0; ct < 16; ++ct) {
      // B = V^T tile: lane col c = cBase0+ct*16+l16 ; K=j contiguous in v[c][n]
      const __bf16* vp = vb + (size_t)(cBase0 + ct * 16 + l16) * N_DIM + jt;
      v16bf bf = load_b32(vp, half);
      oacc[ct] = __builtin_amdgcn_wmma_f32_16x16x32_bf16(false, pA, false, bf, (short)0, oacc[ct], false, false);
    }
  }
  // Normalize by row sum and store transposed for the proj GEMM: oT[b][n][c]
#pragma unroll
  for (int ct = 0; ct < 16; ++ct) {
#pragma unroll
    for (int r = 0; r < 8; ++r) {
      int row = iBase + r + 8 * half;
      int c = cBase0 + ct * 16 + l16;
      float val = oacc[ct][r] / lrow[r];
      oT[((size_t)b * N_DIM + row) * C_DIM + c] = f2bf(val);
    }
  }
}

extern "C" void kernel_launch(void* const* d_in, const int* in_sizes, int n_in,
                              void* d_out, int out_size, void* d_ws, size_t ws_size,
                              hipStream_t stream) {
  (void)in_sizes; (void)n_in; (void)out_size; (void)ws_size;
  const float* x     = (const float*)d_in[0];
  const float* gamma = (const float*)d_in[1];
  const float* beta  = (const float*)d_in[2];
  const float* q_w   = (const float*)d_in[3];
  const float* q_b   = (const float*)d_in[4];
  const float* k_w   = (const float*)d_in[5];
  const float* k_b   = (const float*)d_in[6];
  const float* v_w   = (const float*)d_in[7];
  const float* v_b   = (const float*)d_in[8];
  const float* p_w   = (const float*)d_in[9];
  const float* p_b   = (const float*)d_in[10];

  char* ws = (char*)d_ws;
  size_t off = 0;
  auto take = [&](size_t bytes) -> char* {
    char* p = ws + off;
    off = (off + bytes + 255) & ~(size_t)255;
    return p;
  };
  const size_t actBytes = (size_t)BATCH * N_DIM * C_DIM * sizeof(__bf16);  // 8 MB
  const size_t wBytes = (size_t)C_DIM * C_DIM * sizeof(__bf16);            // 512 KB
  __bf16* hnT = (__bf16*)take(actBytes);
  __bf16* qTb = (__bf16*)take(actBytes);
  __bf16* kTb = (__bf16*)take(actBytes);
  __bf16* vB  = (__bf16*)take(actBytes);
  __bf16* oTb = (__bf16*)take(actBytes);
  __bf16* wq  = (__bf16*)take(wBytes);
  __bf16* wk  = (__bf16*)take(wBytes);
  __bf16* wv  = (__bf16*)take(wBytes);
  __bf16* wp  = (__bf16*)take(wBytes);

  convert_f32_bf16<<<256, 256, 0, stream>>>(q_w, wq, C_DIM * C_DIM);
  convert_f32_bf16<<<256, 256, 0, stream>>>(k_w, wk, C_DIM * C_DIM);
  convert_f32_bf16<<<256, 256, 0, stream>>>(v_w, wv, C_DIM * C_DIM);
  convert_f32_bf16<<<256, 256, 0, stream>>>(p_w, wp, C_DIM * C_DIM);

  groupnorm_kernel<<<dim3(32, BATCH), 256, 0, stream>>>(x, gamma, beta, hnT);

  dim3 ggrid(N_DIM / 128, C_DIM / 64, BATCH);
  gemm_bf16<<<ggrid, 256, 0, stream>>>(wq, hnT, q_b, nullptr, qTb, nullptr, 0);
  gemm_bf16<<<ggrid, 256, 0, stream>>>(wk, hnT, k_b, nullptr, kTb, nullptr, 0);
  gemm_bf16<<<ggrid, 256, 0, stream>>>(wv, hnT, v_b, nullptr, vB,  nullptr, 1);

  attn_kernel<<<dim3(N_DIM / 16, 2, BATCH), 32, 0, stream>>>(qTb, kTb, vB, oTb);

  gemm_bf16<<<ggrid, 256, 0, stream>>>(wp, oTb, p_b, x, nullptr, (float*)d_out, 2);
}
